// LESSR_72129680769668
// MI455X (gfx1250) — compile-verified, matrix operand order checked by reading, not run on hardware
//
#include <hip/hip_runtime.h>
#include <hip/hip_bf16.h>

typedef _Float16 half_t;
typedef __attribute__((ext_vector_type(16))) _Float16 v16h;
typedef __attribute__((ext_vector_type(8)))  _Float16 v8h;
typedef __attribute__((ext_vector_type(8)))  float    v8f;

#define BB 64
#define SS 128
#define DD 64

__device__ __forceinline__ float sigmoidf_(float x) { return 1.0f / (1.0f + __expf(-x)); }

// ---- packed-B position: inverse of ISA 7.12.2 16-bit B-matrix lane/VGPR layout ----
// rel = k % 32;  p = ((rel>>4)<<2)|((rel>>1)&3);  lane = (((rel>>3)&1)<<4)|(n&15)
__device__ __forceinline__ long packB_pos(int k, int n, int tilesN) {
    int kc   = k >> 5;
    int rel  = k & 31;
    int p    = ((rel >> 4) << 2) | ((rel >> 1) & 3);
    int lane = (((rel >> 3) & 1) << 4) | (n & 15);
    int nt   = n >> 4;
    return ((long)(kc * tilesN + nt) * 32 + lane) * 16 + (p << 1) + (rel & 1);
}

// ---------------- pack a row-major [K x N] matrix into WMMA B-fragment order ----------------
template <typename T>
__global__ void k_packB(const T* __restrict__ in, half_t* __restrict__ out,
                        int K, int N, int total) {
    int idx = blockIdx.x * blockDim.x + threadIdx.x;
    if (idx >= total) return;
    int per = K * N;
    int bz  = idx / per;                 // batch (0 for weights)
    int r   = idx - bz * per;
    int k   = r / N;
    int n   = r - k * N;
    out[(long)bz * per + packB_pos(k, n, N >> 4)] = (half_t)in[idx];
}

// ---------------- embedding gather ----------------
__global__ void k_embed(const int* __restrict__ items, const float* __restrict__ emb,
                        float* __restrict__ x, half_t* __restrict__ xh) {
    int t = blockIdx.x;
    int d = threadIdx.x;                 // 64 threads
    float v = emb[(long)items[t] * DD + d];
    x[(long)t * DD + d]  = v;
    xh[(long)t * DD + d] = (half_t)v;
}

// ---------------- EOPA masked neighbor max (no [B,S,S,D] materialization) ----------------
__global__ void k_eopa(const int* __restrict__ A, const int* __restrict__ eo,
                       const unsigned char* __restrict__ mask,
                       const float* __restrict__ x, half_t* __restrict__ nh) {
    int bi = blockIdx.x;
    int b  = bi >> 7;
    int i  = bi & (SS - 1);
    int d  = threadIdx.x;                // 64 threads
    const int* Ab = A  + (long)b * SS * SS;
    const int* eb = eo + (long)b * SS * SS;
    const unsigned char* mb = mask + (long)b * SS;
    const float* xb = x + (long)b * SS * DD;
    float acc = -INFINITY;
    for (int j = 0; j < SS; ++j) {
        int e    = eb[j * SS + i];                    // wave-uniform -> scalar load
        bool c   = (Ab[j * SS + e] == 1) && (mb[j] != 0);
        float xv = xb[j * DD + d];
        acc = fmaxf(acc, c ? xv : 0.0f);
    }
    nh[(long)bi * DD + d] = (half_t)acc;
}

// ---------------- wave-tiled WMMA GEMM, N=64, K compile-time, fused epilogue ----------------
// C[bz] = A[bz] * Bpacked[bz] (+ A2*B2packed) (+ bias[n]) (PReLU) -> f32 and/or f16
template <int K, bool HAS2>
__global__ void k_gemm(const half_t* __restrict__ A, long strideA,
                       const half_t* __restrict__ Bp, long strideB,
                       const half_t* __restrict__ A2, const half_t* __restrict__ B2p,
                       float* __restrict__ C, half_t* __restrict__ C16, long strideC,
                       const float* __restrict__ bias, const float* __restrict__ prelu,
                       int M) {
    constexpr int N = 64;
    constexpr int tilesN = 4;
    int lane = threadIdx.x & 31;
    int wave = threadIdx.x >> 5;
    int tiles = (M >> 4) * tilesN;
    int tile  = blockIdx.x * (blockDim.x >> 5) + wave;
    if (tile >= tiles) return;              // wave-uniform: EXEC all-1s at WMMA
    int mt = tile >> 2;
    int nt = tile & 3;
    long bz = blockIdx.y;

    int m   = (mt << 4) + (lane & 15);
    int hf8 = (lane >> 4) << 3;
    const half_t* arow = A  + bz * strideA + (long)m * K;
    const half_t* bbase = Bp + bz * strideB;

    v8f acc = {};
#pragma unroll
    for (int kc = 0; kc < K / 32; ++kc) {
        v8h lo = *(const v8h*)(arow + kc * 32 + hf8);           // b128
        v8h hi = *(const v8h*)(arow + kc * 32 + 16 + hf8);      // b128
        v16h fa = __builtin_shufflevector(lo, hi, 0, 1, 2, 3, 4, 5, 6, 7,
                                          8, 9, 10, 11, 12, 13, 14, 15);
        v16h fb = *(const v16h*)(bbase + (((long)(kc * tilesN + nt) * 32 + lane) << 4));
        acc = __builtin_amdgcn_wmma_f32_16x16x32_f16(false, fa, false, fb,
                                                     (short)0, acc, false, false);
    }
    if (HAS2) {
        const half_t* arow2 = A2 + bz * strideA + (long)m * K;
#pragma unroll
        for (int kc = 0; kc < K / 32; ++kc) {
            v8h lo = *(const v8h*)(arow2 + kc * 32 + hf8);
            v8h hi = *(const v8h*)(arow2 + kc * 32 + 16 + hf8);
            v16h fa = __builtin_shufflevector(lo, hi, 0, 1, 2, 3, 4, 5, 6, 7,
                                              8, 9, 10, 11, 12, 13, 14, 15);
            v16h fb = *(const v16h*)(B2p + (((long)(kc * tilesN + nt) * 32 + lane) << 4));
            acc = __builtin_amdgcn_wmma_f32_16x16x32_f16(false, fa, false, fb,
                                                         (short)0, acc, false, false);
        }
    }
    int n     = (nt << 4) + (lane & 15);
    int mbase = (mt << 4) + ((lane >> 4) << 3);
    float bia = bias  ? bias[n]  : 0.0f;
    float slp = prelu ? prelu[n] : 1.0f;
#pragma unroll
    for (int r = 0; r < 8; ++r) {
        float v = acc[r] + bia;
        if (prelu) v = (v >= 0.0f) ? v : slp * v;
        long off = bz * strideC + (long)(mbase + r) * N + n;
        if (C)   C[off]   = v;
        if (C16) C16[off] = (half_t)v;
    }
}

// ---------------- SGAT: sigmoid-gated scores + softmax over i, alpha stored transposed f16 ----
__global__ void k_sgat(const int* __restrict__ A, const float* __restrict__ q,
                       const float* __restrict__ k, const float* __restrict__ we,
                       half_t* __restrict__ at16) {
    __shared__ float qs[DD];
    __shared__ float ws[DD];
    __shared__ float red[SS];
    int b = blockIdx.x >> 7;
    int j = blockIdx.x & (SS - 1);
    int i = threadIdx.x;                    // 128 threads = i dimension
    if (i < DD) { qs[i] = q[((long)b * SS + j) * DD + i]; ws[i] = we[i]; }
    __syncthreads();
    bool adj = (A[((long)b * SS + i) * SS + j] == 1);
    const float* kr = k + ((long)b * SS + i) * DD;
    float s = 0.0f;
    for (int d = 0; d < DD; ++d) {
        float t = kr[d] + qs[d];
        t = adj ? t : 0.0f;                 // masked -> sigmoid(0), exact vs reference
        s += ws[d] * sigmoidf_(t);
    }
    red[i] = s; __syncthreads();
    for (int off = SS / 2; off > 0; off >>= 1) {
        if (i < off) red[i] = fmaxf(red[i], red[i + off]);
        __syncthreads();
    }
    float mx = red[0]; __syncthreads();
    float ex = __expf(s - mx);
    red[i] = ex; __syncthreads();
    for (int off = SS / 2; off > 0; off >>= 1) {
        if (i < off) red[i] += red[i + off];
        __syncthreads();
    }
    float alpha = ex / red[0];
    at16[(long)b * SS * SS + (long)j * SS + i] = (half_t)alpha;  // [b, j, i]
}

// ---------------- gather last-node rows ----------------
__global__ void k_last(const float* __restrict__ h2, const int* __restrict__ last,
                       float* __restrict__ xl, half_t* __restrict__ xlh) {
    int b = blockIdx.x;
    int d = threadIdx.x;                    // 64
    float v = h2[((long)b * SS + last[b]) * DD + d];
    xl[(long)b * DD + d]  = v;
    xlh[(long)b * DD + d] = (half_t)v;
}

// ---------------- attention readout ----------------
__global__ void k_readout(const float* __restrict__ xu, const float* __restrict__ xv,
                          const float* __restrict__ wer, const float* __restrict__ h2,
                          const float* __restrict__ prelu3, const float* __restrict__ xl,
                          half_t* __restrict__ cat16) {
    __shared__ float xvs[DD];
    __shared__ float ws[DD];
    __shared__ float red[SS];
    __shared__ float alpha[SS];
    int b = blockIdx.x;
    int i = threadIdx.x;                    // 128 threads = s dimension
    if (i < DD) { xvs[i] = xv[(long)b * DD + i]; ws[i] = wer[i]; }
    __syncthreads();
    const float* xr = xu + ((long)b * SS + i) * DD;
    float s = 0.0f;
    for (int d = 0; d < DD; ++d) s += ws[d] * sigmoidf_(xr[d] + xvs[d]);
    red[i] = s; __syncthreads();
    for (int off = SS / 2; off > 0; off >>= 1) {
        if (i < off) red[i] = fmaxf(red[i], red[i + off]);
        __syncthreads();
    }
    float mx = red[0]; __syncthreads();
    float ex = __expf(s - mx);
    red[i] = ex; __syncthreads();
    for (int off = SS / 2; off > 0; off >>= 1) {
        if (i < off) red[i] += red[i + off];
        __syncthreads();
    }
    alpha[i] = ex / red[0];
    __syncthreads();
    if (i < DD) {
        float acc = 0.0f;
        for (int t = 0; t < SS; ++t) acc += alpha[t] * h2[((long)b * SS + t) * DD + i];
        float o = (acc >= 0.0f) ? acc : prelu3[i] * acc;
        cat16[(long)b * 2 * DD + i]      = (half_t)o;
        cat16[(long)b * 2 * DD + DD + i] = (half_t)xl[(long)b * DD + i];
    }
}

// ---------------- host-side orchestration ----------------
static inline int gemm_grid(int M) { return ((M >> 4) * 4 + 7) / 8; }

extern "C" void kernel_launch(void* const* d_in, const int* in_sizes, int n_in,
                              void* d_out, int out_size, void* d_ws, size_t ws_size,
                              hipStream_t stream) {
    (void)in_sizes; (void)n_in; (void)out_size; (void)ws_size;
    const int*           items  = (const int*)d_in[0];
    const int*           A      = (const int*)d_in[1];
    const int*           eo     = (const int*)d_in[2];
    const int*           last   = (const int*)d_in[3];
    const unsigned char* mask   = (const unsigned char*)d_in[4];
    const float*         emb    = (const float*)d_in[5];
    const float*         Wself  = (const float*)d_in[6];
    const float*         Wneigh = (const float*)d_in[7];
    const float*         prelu1 = (const float*)d_in[8];
    const float*         Wq     = (const float*)d_in[9];
    const float*         bq     = (const float*)d_in[10];
    const float*         Wk     = (const float*)d_in[11];
    const float*         Wv     = (const float*)d_in[12];
    const float*         we     = (const float*)d_in[13];
    const float*         prelu2 = (const float*)d_in[14];
    const float*         Wu     = (const float*)d_in[15];
    const float*         bu     = (const float*)d_in[16];
    const float*         Wvr    = (const float*)d_in[17];
    const float*         wer    = (const float*)d_in[18];
    const float*         prelu3 = (const float*)d_in[19];
    const float*         Wsr    = (const float*)d_in[20];
    float* sr = (float*)d_out;

    const int BS = BB * SS;                 // 8192
    char* wp = (char*)d_ws;
    auto alloc = [&](size_t bytes) -> void* {
        void* p = (void*)wp;
        wp += (bytes + 255) & ~(size_t)255;
        return p;
    };
    float*  x     = (float*)alloc((size_t)BS * DD * 4);
    half_t* xh    = (half_t*)alloc((size_t)BS * DD * 2);
    half_t* nh    = (half_t*)alloc((size_t)BS * DD * 2);
    half_t* hh    = (half_t*)alloc((size_t)BS * DD * 2);
    float*  qf    = (float*)alloc((size_t)BS * DD * 4);
    float*  kf    = (float*)alloc((size_t)BS * DD * 4);
    half_t* vtmp  = (half_t*)alloc((size_t)BS * DD * 2);   // v, row-major f16
    half_t* vhp   = (half_t*)alloc((size_t)BS * DD * 2);   // v, B-fragment packed per batch
    half_t* at16  = (half_t*)alloc((size_t)BB * SS * SS * 2);
    float*  h2f   = (float*)alloc((size_t)BS * DD * 4);
    half_t* h2h   = (half_t*)alloc((size_t)BS * DD * 2);
    float*  xuf   = (float*)alloc((size_t)BS * DD * 4);
    float*  xlf   = (float*)alloc((size_t)BB * DD * 4);
    half_t* xlh   = (half_t*)alloc((size_t)BB * DD * 2);
    float*  xvf   = (float*)alloc((size_t)BB * DD * 4);
    half_t* cat16 = (half_t*)alloc((size_t)BB * 2 * DD * 2);
    half_t* wself16  = (half_t*)alloc((size_t)DD * DD * 2);
    half_t* wneigh16 = (half_t*)alloc((size_t)DD * DD * 2);
    half_t* wq16     = (half_t*)alloc((size_t)DD * DD * 2);
    half_t* wk16     = (half_t*)alloc((size_t)DD * DD * 2);
    half_t* wv16     = (half_t*)alloc((size_t)DD * DD * 2);
    half_t* wu16     = (half_t*)alloc((size_t)DD * DD * 2);
    half_t* wvr16    = (half_t*)alloc((size_t)DD * DD * 2);
    half_t* wsr16    = (half_t*)alloc((size_t)2 * DD * DD * 2);

    // weight packs (f32 -> f16, B-fragment order)
    int nw = DD * DD;
    k_packB<float><<<(nw + 255) / 256, 256, 0, stream>>>(Wself,  wself16,  DD, DD, nw);
    k_packB<float><<<(nw + 255) / 256, 256, 0, stream>>>(Wneigh, wneigh16, DD, DD, nw);
    k_packB<float><<<(nw + 255) / 256, 256, 0, stream>>>(Wq,     wq16,     DD, DD, nw);
    k_packB<float><<<(nw + 255) / 256, 256, 0, stream>>>(Wk,     wk16,     DD, DD, nw);
    k_packB<float><<<(nw + 255) / 256, 256, 0, stream>>>(Wv,     wv16,     DD, DD, nw);
    k_packB<float><<<(nw + 255) / 256, 256, 0, stream>>>(Wu,     wu16,     DD, DD, nw);
    k_packB<float><<<(nw + 255) / 256, 256, 0, stream>>>(Wvr,    wvr16,    DD, DD, nw);
    k_packB<float><<<(2 * nw + 255) / 256, 256, 0, stream>>>(Wsr, wsr16, 2 * DD, DD, 2 * nw);

    // x = emb[items]; EOPA neighbor max
    k_embed<<<BS, DD, 0, stream>>>(items, emb, x, xh);
    k_eopa<<<BS, DD, 0, stream>>>(A, eo, mask, x, nh);

    // h = prelu1(x@W_self + neigh@W_neigh)  -> f16
    k_gemm<DD, true><<<dim3(gemm_grid(BS), 1), 256, 0, stream>>>(
        xh, 0, wself16, 0, nh, wneigh16, nullptr, hh, 0, nullptr, prelu1, BS);
    // q = h@Wq + bq ; k = h@Wk ; v = h@Wv
    k_gemm<DD, false><<<dim3(gemm_grid(BS), 1), 256, 0, stream>>>(
        hh, 0, wq16, 0, nullptr, nullptr, qf, nullptr, 0, bq, nullptr, BS);
    k_gemm<DD, false><<<dim3(gemm_grid(BS), 1), 256, 0, stream>>>(
        hh, 0, wk16, 0, nullptr, nullptr, kf, nullptr, 0, nullptr, nullptr, BS);
    k_gemm<DD, false><<<dim3(gemm_grid(BS), 1), 256, 0, stream>>>(
        hh, 0, wv16, 0, nullptr, nullptr, nullptr, vtmp, 0, nullptr, nullptr, BS);

    // pack v into per-batch B-fragment order for the h2 contraction
    k_packB<half_t><<<(BS * DD + 255) / 256, 256, 0, stream>>>(vtmp, vhp, SS, DD, BS * DD);

    // attention scores + softmax over i
    k_sgat<<<BS, SS, 0, stream>>>(A, qf, kf, we, at16);

    // h2[b] = prelu2(alphaT[b] @ v[b]) — batched, 64 batches
    k_gemm<SS, false><<<dim3(gemm_grid(SS), BB), 256, 0, stream>>>(
        at16, (long)SS * SS, vhp, (long)SS * DD, nullptr, nullptr,
        h2f, h2h, (long)SS * DD, nullptr, prelu2, SS);

    // xu = h2@Wu + bu ; xlast ; xv = xlast@Wvr
    k_gemm<DD, false><<<dim3(gemm_grid(BS), 1), 256, 0, stream>>>(
        h2h, 0, wu16, 0, nullptr, nullptr, xuf, nullptr, 0, bu, nullptr, BS);
    k_last<<<BB, DD, 0, stream>>>(h2f, last, xlf, xlh);
    k_gemm<DD, false><<<dim3(gemm_grid(BB), 1), 256, 0, stream>>>(
        xlh, 0, wvr16, 0, nullptr, nullptr, xvf, nullptr, 0, nullptr, nullptr, BB);

    // readout -> cat16
    k_readout<<<BB, SS, 0, stream>>>(xuf, xvf, wer, h2f, prelu3, xlf, cat16);

    // sr = cat @ W_sr -> d_out (f32)
    k_gemm<2 * DD, false><<<dim3(gemm_grid(BB), 1), 256, 0, stream>>>(
        cat16, 0, wsr16, 0, nullptr, nullptr, sr, nullptr, 0, nullptr, nullptr, BB);
}
